// cylinder_fea_33689723470415
// MI455X (gfx1250) — compile-verified, matrix operand order checked by reading, not run on hardware
//
#include <hip/hip_runtime.h>

// ---------------- problem constants ----------------
#define BATCH   2
#define NPTS    65536
#define MTOT    131072          // B*N
#define FEA     9
#define COMPRE  16
#define GXD     240
#define GYD     180
#define GZD     32
#define KEYSP   (BATCH*GXD*GYD*GZD)   // 2,764,800
#define MAXU    MTOT
#define SCAN_BLK_ELEMS 1024
#define NBLK_SCAN ((KEYSP + SCAN_BLK_ELEMS - 1) / SCAN_BLK_ELEMS)  // 2700
#define ENC_NEG_INF 0x00800000u       // fenc(-FLT_MAX)
#define INV_NPTS (1.0f/65536.0f)

// ---------------- WMMA types ----------------
typedef __attribute__((ext_vector_type(16))) __bf16 v16bf;
typedef __attribute__((ext_vector_type(8)))  __bf16 bf16x8;   // 16B, align 16
typedef __attribute__((ext_vector_type(8)))  float  v8f;
typedef __attribute__((ext_vector_type(4)))  int    v4i_t;

// ---------------- async global->LDS copy (CDNA5), guarded ----------------
#if defined(__HIP_DEVICE_COMPILE__) && defined(__has_builtin)
#if __has_builtin(__builtin_amdgcn_global_load_async_to_lds_b128) && \
    __has_builtin(__builtin_amdgcn_s_wait_asynccnt)
#define HAVE_ASYNC_LDS 1
#endif
#endif

#ifdef HAVE_ASYNC_LDS
typedef __attribute__((address_space(1))) v4i_t* gptr_v4i;   // prints as __device__ int4*
typedef __attribute__((address_space(3))) v4i_t* lptr_v4i;   // prints as __shared__ int4*
__device__ __forceinline__ void async_copy16(const void* g, void* l) {
    // int round-trips avoid const/AS-cast issues; flat->LDS truncation is valid:
    // ISA 10.2: the LDS address is the low 32 bits of the flat address (aperture tag is in the high 32 bits).
    gptr_v4i gp = (gptr_v4i)(unsigned long long)g;
    lptr_v4i lp = (lptr_v4i)(unsigned)(unsigned long long)l;
    __builtin_amdgcn_global_load_async_to_lds_b128(gp, lp, 0, 0);
}
__device__ __forceinline__ void async_wait() { __builtin_amdgcn_s_wait_asynccnt(0); }
#else
__device__ __forceinline__ void async_copy16(const void* g, void* l) {
    *(bf16x8*)l = *(const bf16x8*)g;
}
__device__ __forceinline__ void async_wait() {}
#endif

// monotone float <-> u32 encoding for atomicMax-based segment_max
__device__ __forceinline__ unsigned fenc(float f) {
    unsigned u = __float_as_uint(f);
    return (u >> 31) ? ~u : (u | 0x80000000u);
}
__device__ __forceinline__ float fdec(unsigned u) {
    return (u >> 31) ? __uint_as_float(u & 0x7fffffffu) : __uint_as_float(~u);
}

// ---------------- utility: fill u32 ----------------
__global__ void k_fill_u32(unsigned* __restrict__ p, unsigned val, int n) {
    for (int i = blockIdx.x * blockDim.x + threadIdx.x; i < n; i += gridDim.x * blockDim.x)
        p[i] = val;
}

// ---------------- keys + occupancy ----------------
__global__ void k_keys(const int* __restrict__ xy, int* __restrict__ keys,
                       unsigned* __restrict__ occ) {
    int p = blockIdx.x * blockDim.x + threadIdx.x;
    if (p >= MTOT) return;
    int b = p >> 16;
    int x = xy[p * 3 + 0], y = xy[p * 3 + 1], z = xy[p * 3 + 2];
    int key = ((b * GXD + x) * GYD + y) * GZD + z;
    keys[p] = key;
    occ[key] = 1u;
}

// ---------------- occupancy scan (dense-key unique) ----------------
__global__ __launch_bounds__(256)
void k_scan_local(const unsigned* __restrict__ occ, unsigned* __restrict__ rank,
                  unsigned* __restrict__ bsum, int n) {
    __shared__ unsigned s[256];
    const int tid = threadIdx.x;
    const int base = blockIdx.x * SCAN_BLK_ELEMS;
    unsigned v[4], tsum = 0;
#pragma unroll
    for (int i = 0; i < 4; ++i) {
        int idx = base + tid * 4 + i;
        v[i] = (idx < n) ? occ[idx] : 0u;
        tsum += v[i];
    }
    s[tid] = tsum; __syncthreads();
    for (int off = 1; off < 256; off <<= 1) {
        unsigned t = (tid >= off) ? s[tid - off] : 0u;
        __syncthreads();
        s[tid] += t;
        __syncthreads();
    }
    unsigned run = s[tid] - tsum;
    if (tid == 255) bsum[blockIdx.x] = s[255];
#pragma unroll
    for (int i = 0; i < 4; ++i) {
        int idx = base + tid * 4 + i;
        if (idx < n) rank[idx] = run;
        run += v[i];
    }
}

__global__ __launch_bounds__(256)
void k_scan_bsum(unsigned* __restrict__ bsum, int nb, unsigned* __restrict__ nvox) {
    __shared__ unsigned s[256];
    __shared__ unsigned carry;
    const int tid = threadIdx.x;
    if (tid == 0) carry = 0;
    __syncthreads();
    for (int base = 0; base < nb; base += 256) {
        int i = base + tid;
        unsigned v = (i < nb) ? bsum[i] : 0u;
        s[tid] = v; __syncthreads();
        for (int off = 1; off < 256; off <<= 1) {
            unsigned t = (tid >= off) ? s[tid - off] : 0u;
            __syncthreads();
            s[tid] += t;
            __syncthreads();
        }
        if (i < nb) bsum[i] = s[tid] - v + carry;
        __syncthreads();
        if (tid == 0) carry += s[255];
        __syncthreads();
    }
    if (tid == 0) *nvox = carry;
}

__global__ void k_scan_emit(const unsigned* __restrict__ occ, unsigned* __restrict__ rank,
                            const unsigned* __restrict__ bsum, int* __restrict__ uk, int n) {
    int i = blockIdx.x * blockDim.x + threadIdx.x;
    if (i >= n) return;
    unsigned r = rank[i] + bsum[i >> 10];
    rank[i] = r;
    if (occ[i]) uk[r] = i;
}

__global__ void k_inv(const int* __restrict__ keys, const unsigned* __restrict__ rank,
                      int* __restrict__ inv) {
    int p = blockIdx.x * blockDim.x + threadIdx.x;
    if (p < MTOT) inv[p] = (int)rank[keys[p]];
}

// ---------------- per-batch column stats (sum, sumsq) ----------------
__global__ __launch_bounds__(256)
void k_colstats(const float* __restrict__ X, int C, float* __restrict__ stats) {
    __shared__ float sh[256], sh2[256];
    const int tid = threadIdx.x;
    const int c = blockIdx.y;
    const int r0 = blockIdx.x * 4096;
    const int b = r0 >> 16;
    float s = 0.f, q = 0.f;
    for (int r = r0 + tid; r < r0 + 4096; r += 256) {
        float v = X[(size_t)r * C + c];
        s += v; q += v * v;
    }
    sh[tid] = s; sh2[tid] = q; __syncthreads();
    for (int off = 128; off > 0; off >>= 1) {
        if (tid < off) { sh[tid] += sh[tid + off]; sh2[tid] += sh2[tid + off]; }
        __syncthreads();
    }
    if (tid == 0) {
        atomicAdd(&stats[b * C + c], sh[0]);
        atomicAdd(&stats[(2 + b) * C + c], sh2[0]);
    }
}

// ---------------- weight convert: Wt[n][k] = bf16(W[k][n]), zero-padded ----------------
__global__ void k_wconv(const float* __restrict__ W, __bf16* __restrict__ Wt,
                        int K, int N, int Kp, int Np) {
    int idx = blockIdx.x * blockDim.x + threadIdx.x;
    if (idx >= Np * Kp) return;
    int n = idx / Kp, k = idx % Kp;
    float v = (n < N && k < K) ? W[(size_t)k * N + n] : 0.f;
    Wt[idx] = (__bf16)v;
}

// ---------------- bf16 WMMA GEMM, 128x64 block tile, 2x2 tiles per wave ----------------
// in_mode: 1 = BN, 2 = BN+ReLU (X f32), 3 = A already bf16 (Xb, async-staged)
// out_mode: 0 = plain, 2 = ReLU + zero rows >= *nvox_ptr
#define TM 128
#define TN 64
#define TK 32
#define LDP 40   // padded LDS stride: 80 B = 5 x 16 B

__global__ __launch_bounds__(256)
void k_gemm(const float* __restrict__ X, const __bf16* __restrict__ Xb,
            const __bf16* __restrict__ Wt, int Kp,
            const float* __restrict__ bias, float* __restrict__ Y,
            int Mrows, int K, int N,
            int in_mode, const float* __restrict__ stats,
            const float* __restrict__ bng, const float* __restrict__ bnb,
            int out_mode, const unsigned* __restrict__ nvox_ptr) {
    __shared__ __bf16 As[TM][LDP];
    __shared__ __bf16 Bs[TN][LDP];
    const int tid = threadIdx.x;
    const int rowBase = blockIdx.x * TM;
    const int colBase = blockIdx.y * TN;
    const int lane = tid & 31;
    const int wave = tid >> 5;
    const int wr = wave & 3;     // 4 row groups of 32 rows
    const int wc = wave >> 2;    // 2 col groups of 32 cols
    v8f acc00 = {0.f,0.f,0.f,0.f,0.f,0.f,0.f,0.f};
    v8f acc01 = acc00, acc10 = acc00, acc11 = acc00;

    const int ksteps = (K + TK - 1) / TK;
    for (int ks = 0; ks < ksteps; ++ks) {
        const int k0 = ks * TK;
        // ---- stage A tile (128 x 32 bf16) ----
        if (in_mode == 3) {
            // raw bf16: 512 x 16B chunks, 2 per thread, async copy
#pragma unroll
            for (int i = 0; i < 2; ++i) {
                int e = tid * 2 + i;
                int r = e >> 2, c = e & 3;          // 4 chunks per 64B row
                async_copy16(Xb + (size_t)(rowBase + r) * K + k0 + c * 8,
                             &As[r][c * 8]);
            }
        } else {
            if (ks + 1 < ksteps && (tid & 63) == 0)
                __builtin_prefetch(X + (size_t)(rowBase + tid / 2) * K + k0 + TK, 0, 1);
#pragma unroll
            for (int i = 0; i < 16; ++i) {
                int e = tid * 16 + i;
                int r = e >> 5, kk = e & 31;
                int gk = k0 + kk, grow = rowBase + r;
                float v = 0.f;
                if (gk < K) {
                    v = X[(size_t)grow * K + gk];
                    int b = grow >> 16;
                    float mu  = stats[b * K + gk] * INV_NPTS;
                    float var = stats[(2 + b) * K + gk] * INV_NPTS - mu * mu;
                    float istd = rsqrtf(var + 1e-5f);
                    v = (v - mu) * istd * bng[gk] + bnb[gk];
                    if (in_mode == 2) v = fmaxf(v, 0.f);
                }
                As[r][kk] = (__bf16)v;
            }
        }
        // ---- stage B tile (64 x 32 bf16, pre-transposed & padded): async ----
        {
            int n = tid >> 2, c = tid & 3;
            async_copy16(Wt + (size_t)(colBase + n) * Kp + k0 + c * 8, &Bs[n][c * 8]);
        }
        async_wait();
        __syncthreads();

        // ---- fragments: ISA 16-bit layout (lanes 0-15 Kbase 0, 16-31 Kbase 8) ----
        const int rl = lane & 15;
        const int kb = (lane < 16) ? 0 : 8;
        v16bf a0, a1, b0, b1;
        {
            const __bf16* ap = &As[wr * 32 + rl][0];
            bf16x8 x0 = *(const bf16x8*)(ap + kb);
            bf16x8 x1 = *(const bf16x8*)(ap + kb + 16);
            const __bf16* aq = &As[wr * 32 + 16 + rl][0];
            bf16x8 y0 = *(const bf16x8*)(aq + kb);
            bf16x8 y1 = *(const bf16x8*)(aq + kb + 16);
            const __bf16* bp = &Bs[wc * 32 + rl][0];
            bf16x8 u0 = *(const bf16x8*)(bp + kb);
            bf16x8 u1 = *(const bf16x8*)(bp + kb + 16);
            const __bf16* bq = &Bs[wc * 32 + 16 + rl][0];
            bf16x8 w0 = *(const bf16x8*)(bq + kb);
            bf16x8 w1 = *(const bf16x8*)(bq + kb + 16);
#pragma unroll
            for (int i = 0; i < 8; ++i) {
                a0[i] = x0[i]; a0[8 + i] = x1[i];
                a1[i] = y0[i]; a1[8 + i] = y1[i];
                b0[i] = u0[i]; b0[8 + i] = u1[i];
                b1[i] = w0[i]; b1[8 + i] = w1[i];
            }
        }
        acc00 = __builtin_amdgcn_wmma_f32_16x16x32_bf16(false, a0, false, b0, (short)0, acc00, false, false);
        acc01 = __builtin_amdgcn_wmma_f32_16x16x32_bf16(false, a0, false, b1, (short)0, acc01, false, false);
        acc10 = __builtin_amdgcn_wmma_f32_16x16x32_bf16(false, a1, false, b0, (short)0, acc10, false, false);
        acc11 = __builtin_amdgcn_wmma_f32_16x16x32_bf16(false, a1, false, b1, (short)0, acc11, false, false);
        __syncthreads();
    }

    // ---- epilogue: C layout -> n = lane&15, m = j + 8*(lane>=16) ----
    unsigned nv = nvox_ptr ? *nvox_ptr : 0u;
    const int cl = lane & 15;
    const int rhi = (lane >> 4) << 3;
#pragma unroll
    for (int mt = 0; mt < 2; ++mt) {
#pragma unroll
        for (int nt = 0; nt < 2; ++nt) {
            const v8f* accp = (mt == 0) ? (nt == 0 ? &acc00 : &acc01)
                                        : (nt == 0 ? &acc10 : &acc11);
            int gcol  = colBase + wc * 32 + nt * 16 + cl;
            int mrow0 = rowBase + wr * 32 + mt * 16 + rhi;
            if (gcol < N) {
                float bvv = bias[gcol];
#pragma unroll
                for (int j = 0; j < 8; ++j) {
                    int grow = mrow0 + j;
                    float o = (*accp)[j] + bvv;
                    if (out_mode == 2) {
                        o = fmaxf(o, 0.f);
                        if ((unsigned)grow >= nv) o = 0.f;
                    }
                    Y[(size_t)grow * N + gcol] = o;
                }
            }
        }
    }
}

// ---------------- scatter max (segment_max over voxels) ----------------
__global__ void k_scatter(const float* __restrict__ mlp, const float* __restrict__ seg,
                          const float* __restrict__ pix, const int* __restrict__ inv,
                          unsigned* __restrict__ pm, unsigned* __restrict__ ps,
                          unsigned* __restrict__ pp) {
    const int total = MTOT * 256;
    for (int idx = blockIdx.x * blockDim.x + threadIdx.x; idx < total;
         idx += gridDim.x * blockDim.x) {
        int p = idx >> 8, c = idx & 255;
        unsigned base = (unsigned)inv[p] * 256u + c;
        atomicMax(&pm[base], fenc(mlp[idx]));
        atomicMax(&ps[base], fenc(seg[idx]));
        atomicMax(&pp[base], fenc(pix[idx]));
    }
}

// ---------------- decode pooled -> bf16 + seg-nonzero flag ----------------
__global__ void k_decode(const unsigned* __restrict__ pm, const unsigned* __restrict__ ps,
                         const unsigned* __restrict__ pp, __bf16* __restrict__ pmh,
                         __bf16* __restrict__ psh, __bf16* __restrict__ pph,
                         unsigned* __restrict__ flag, const unsigned* __restrict__ nvp) {
    const unsigned nv = *nvp;
    const int total = MAXU * 256;
    for (int idx = blockIdx.x * blockDim.x + threadIdx.x; idx < total;
         idx += gridDim.x * blockDim.x) {
        int vox = idx >> 8;
        bool valid = (unsigned)vox < nv;
        float fm = valid ? fdec(pm[idx]) : 0.f;
        float fs = valid ? fdec(ps[idx]) : 0.f;
        float fp = valid ? fdec(pp[idx]) : 0.f;
        pmh[idx] = (__bf16)fm;
        psh[idx] = (__bf16)fs;
        pph[idx] = (__bf16)fp;
        if (fs != 0.f) atomicOr(&flag[vox], 1u);
    }
}

// ---------------- fuse -> bf16 A for the compress GEMM ----------------
__global__ void k_fuse(const float* __restrict__ v128, const float* __restrict__ s128,
                       const float* __restrict__ p128, const unsigned* __restrict__ flag,
                       __bf16* __restrict__ fused) {
    const int total = MAXU * 128;
    for (int idx = blockIdx.x * blockDim.x + threadIdx.x; idx < total;
         idx += gridDim.x * blockDim.x) {
        int vox = idx >> 7;
        float f = v128[idx];
        if (flag[vox]) f += s128[idx] + p128[idx];
        fused[idx] = (__bf16)f;
    }
}

// ---------------- decode unique keys to (b, gx, gy, gz) ----------------
__global__ void k_unq(const int* __restrict__ uk, const unsigned* __restrict__ nvp,
                      int* __restrict__ out) {
    int vox = blockIdx.x * blockDim.x + threadIdx.x;
    if (vox >= MAXU) return;
    unsigned nv = *nvp;
    int* o = out + vox * 4;
    if ((unsigned)vox < nv) {
        int k = uk[vox];
        int b = k / (GXD * GYD * GZD);
        int rem = k % (GXD * GYD * GZD);
        o[0] = b;
        o[1] = rem / (GYD * GZD);
        o[2] = (rem / GZD) % GYD;
        o[3] = rem % GZD;
    } else {
        o[0] = -1; o[1] = -1; o[2] = -1; o[3] = -1;
    }
}

// ==================================================================
extern "C" void kernel_launch(void* const* d_in, const int* in_sizes, int n_in,
                              void* d_out, int out_size, void* d_ws, size_t ws_size,
                              hipStream_t stream) {
    (void)in_sizes; (void)n_in; (void)out_size; (void)ws_size;
    const float* pt_fea = (const float*)d_in[0];
    const int*   xy_ind = (const int*)d_in[1];
    const float* segfea = (const float*)d_in[2];
    const float* pixfea = (const float*)d_in[3];
    const float *bn0g = (const float*)d_in[4],  *bn0b = (const float*)d_in[5];
    const float *l0w  = (const float*)d_in[6],  *l0b  = (const float*)d_in[7];
    const float *bn1g = (const float*)d_in[8],  *bn1b = (const float*)d_in[9];
    const float *l1w  = (const float*)d_in[10], *l1b  = (const float*)d_in[11];
    const float *bn2g = (const float*)d_in[12], *bn2b = (const float*)d_in[13];
    const float *l2w  = (const float*)d_in[14], *l2b  = (const float*)d_in[15];
    const float *bn3g = (const float*)d_in[16], *bn3b = (const float*)d_in[17];
    const float *l3w  = (const float*)d_in[18], *l3b  = (const float*)d_in[19];
    const float *t128w = (const float*)d_in[20], *t128b = (const float*)d_in[21];
    const float *s128w = (const float*)d_in[22], *s128b = (const float*)d_in[23];
    const float *p128w = (const float*)d_in[24], *p128b = (const float*)d_in[25];
    const float *cw    = (const float*)d_in[26], *cb    = (const float*)d_in[27];

    char* ws = (char*)d_ws;
    size_t off = 0;
    auto carve = [&](size_t bytes) -> char* {
        char* p = ws + off;
        off += (bytes + 255) & ~(size_t)255;
        return p;
    };
    int*      keys  = (int*)     carve((size_t)MTOT * 4);
    int*      inv   = (int*)     carve((size_t)MTOT * 4);
    unsigned* occ   = (unsigned*)carve((size_t)KEYSP * 4);
    unsigned* rank  = (unsigned*)carve((size_t)KEYSP * 4);
    unsigned* bsum  = (unsigned*)carve((size_t)NBLK_SCAN * 4);
    unsigned* nvox  = (unsigned*)carve(4);
    int*      uk    = (int*)     carve((size_t)MAXU * 4);
    unsigned* flag  = (unsigned*)carve((size_t)MAXU * 4);
    float* stats_f  = (float*)   carve(4 * 256 * 4);
    float* stats_h1 = (float*)   carve(4 * 256 * 4);
    float* stats_h2 = (float*)   carve(4 * 256 * 4);
    float* stats_h3 = (float*)   carve(4 * 256 * 4);
    // pre-transposed bf16 weights [Np][Kp]
    __bf16* wt_l0 = (__bf16*)carve(64  * 32  * 2);
    __bf16* wt_l1 = (__bf16*)carve(128 * 64  * 2);
    __bf16* wt_l2 = (__bf16*)carve(256 * 128 * 2);
    __bf16* wt_l3 = (__bf16*)carve(256 * 256 * 2);
    __bf16* wt_t  = (__bf16*)carve(128 * 256 * 2);
    __bf16* wt_s  = (__bf16*)carve(128 * 256 * 2);
    __bf16* wt_p  = (__bf16*)carve(128 * 256 * 2);
    __bf16* wt_c  = (__bf16*)carve(64  * 128 * 2);
    char* regA = carve((size_t)MTOT * (64 + 128 + 256 + 256) * 4);  // h1|h2|h3|mlpout
    float* h1  = (float*)regA;
    float* h2  = (float*)(regA + (size_t)MTOT * 64 * 4);
    float* h3  = (float*)(regA + (size_t)MTOT * (64 + 128) * 4);
    float* mlp = (float*)(regA + (size_t)MTOT * (64 + 128 + 256) * 4);
    unsigned* pm = (unsigned*)carve((size_t)MAXU * 256 * 4);
    unsigned* ps = (unsigned*)carve((size_t)MAXU * 256 * 4);
    unsigned* pp = (unsigned*)carve((size_t)MAXU * 256 * 4);
    __bf16* pmh = (__bf16*)carve((size_t)MAXU * 256 * 2);
    __bf16* psh = (__bf16*)carve((size_t)MAXU * 256 * 2);
    __bf16* pph = (__bf16*)carve((size_t)MAXU * 256 * 2);
    // after scatter, regA is dead: reuse for the voxel head
    float*  v128  = (float*)regA;
    float*  s128  = (float*)(regA + (size_t)MAXU * 128 * 4);
    float*  p128  = (float*)(regA + (size_t)MAXU * 128 * 4 * 2);
    __bf16* fused = (__bf16*)(regA + (size_t)MAXU * 128 * 4 * 3);

    int*   unq_out = (int*)d_out;                       // MAXU x 4 int32
    float* out_f   = (float*)d_out + (size_t)MAXU * 4;  // MAXU x 16 float

    const dim3 blk(256);

    // ---- init ----
    k_fill_u32<<<dim3(2048), blk, 0, stream>>>(occ, 0u, KEYSP);
    k_fill_u32<<<dim3(512), blk, 0, stream>>>((unsigned*)uk, 0xFFFFFFFFu, MAXU);
    k_fill_u32<<<dim3(512), blk, 0, stream>>>(flag, 0u, MAXU);
    k_fill_u32<<<dim3(4), blk, 0, stream>>>((unsigned*)stats_f, 0u, 4 * 256 * 4);
    k_fill_u32<<<dim3(8192), blk, 0, stream>>>(pm, ENC_NEG_INF, MAXU * 256);
    k_fill_u32<<<dim3(8192), blk, 0, stream>>>(ps, ENC_NEG_INF, MAXU * 256);
    k_fill_u32<<<dim3(8192), blk, 0, stream>>>(pp, ENC_NEG_INF, MAXU * 256);

    // ---- weight pre-transpose/convert ----
    k_wconv<<<dim3((64*32 +255)/256), blk, 0, stream>>>(l0w,  wt_l0, FEA, 64,  32,  64);
    k_wconv<<<dim3((128*64 +255)/256), blk, 0, stream>>>(l1w,  wt_l1, 64,  128, 64,  128);
    k_wconv<<<dim3((256*128+255)/256), blk, 0, stream>>>(l2w,  wt_l2, 128, 256, 128, 256);
    k_wconv<<<dim3((256*256+255)/256), blk, 0, stream>>>(l3w,  wt_l3, 256, 256, 256, 256);
    k_wconv<<<dim3((128*256+255)/256), blk, 0, stream>>>(t128w, wt_t, 256, 128, 256, 128);
    k_wconv<<<dim3((128*256+255)/256), blk, 0, stream>>>(s128w, wt_s, 256, 128, 256, 128);
    k_wconv<<<dim3((128*256+255)/256), blk, 0, stream>>>(p128w, wt_p, 256, 128, 256, 128);
    k_wconv<<<dim3((64*128 +255)/256), blk, 0, stream>>>(cw,    wt_c, 128, 16,  128, 64);

    // ---- keys / unique via occupancy scan ----
    k_keys<<<dim3(MTOT / 256), blk, 0, stream>>>(xy_ind, keys, occ);
    k_scan_local<<<dim3(NBLK_SCAN), blk, 0, stream>>>(occ, rank, bsum, KEYSP);
    k_scan_bsum<<<dim3(1), blk, 0, stream>>>(bsum, NBLK_SCAN, nvox);
    k_scan_emit<<<dim3((KEYSP + 255) / 256), blk, 0, stream>>>(occ, rank, bsum, uk, KEYSP);
    k_inv<<<dim3(MTOT / 256), blk, 0, stream>>>(keys, rank, inv);

    // ---- per-point MLP (bf16 WMMA GEMMs, BN fused into A staging) ----
    k_colstats<<<dim3(32, FEA), blk, 0, stream>>>(pt_fea, FEA, stats_f);
    k_gemm<<<dim3(MTOT / TM, 1), blk, 0, stream>>>(
        pt_fea, nullptr, wt_l0, 32, l0b, h1, MTOT, FEA, 64, 1, stats_f, bn0g, bn0b, 0, nullptr);
    k_colstats<<<dim3(32, 64), blk, 0, stream>>>(h1, 64, stats_h1);
    k_gemm<<<dim3(MTOT / TM, 2), blk, 0, stream>>>(
        h1, nullptr, wt_l1, 64, l1b, h2, MTOT, 64, 128, 2, stats_h1, bn1g, bn1b, 0, nullptr);
    k_colstats<<<dim3(32, 128), blk, 0, stream>>>(h2, 128, stats_h2);
    k_gemm<<<dim3(MTOT / TM, 4), blk, 0, stream>>>(
        h2, nullptr, wt_l2, 128, l2b, h3, MTOT, 128, 256, 2, stats_h2, bn2g, bn2b, 0, nullptr);
    k_colstats<<<dim3(32, 256), blk, 0, stream>>>(h3, 256, stats_h3);
    k_gemm<<<dim3(MTOT / TM, 4), blk, 0, stream>>>(
        h3, nullptr, wt_l3, 256, l3b, mlp, MTOT, 256, 256, 2, stats_h3, bn3g, bn3b, 0, nullptr);

    // ---- voxel pooling (segment_max) ----
    k_scatter<<<dim3(MTOT), blk, 0, stream>>>(mlp, segfea, pixfea, inv, pm, ps, pp);
    k_decode<<<dim3(MAXU), blk, 0, stream>>>(pm, ps, pp, pmh, psh, pph, flag, nvox);

    // ---- voxel head GEMMs (bf16 A, fully async-staged) ----
    k_gemm<<<dim3(MAXU / TM, 2), blk, 0, stream>>>(
        nullptr, pmh, wt_t, 256, t128b, v128, MAXU, 256, 128, 3, nullptr, nullptr, nullptr, 0, nullptr);
    k_gemm<<<dim3(MAXU / TM, 2), blk, 0, stream>>>(
        nullptr, psh, wt_s, 256, s128b, s128, MAXU, 256, 128, 3, nullptr, nullptr, nullptr, 0, nullptr);
    k_gemm<<<dim3(MAXU / TM, 2), blk, 0, stream>>>(
        nullptr, pph, wt_p, 256, p128b, p128, MAXU, 256, 128, 3, nullptr, nullptr, nullptr, 0, nullptr);
    k_fuse<<<dim3(MAXU / 2), blk, 0, stream>>>(v128, s128, p128, flag, fused);
    k_gemm<<<dim3(MAXU / TM, 1), blk, 0, stream>>>(
        nullptr, fused, wt_c, 128, cb, out_f, MAXU, 128, COMPRE, 3, nullptr, nullptr, nullptr, 2, nvox);

    // ---- decode unique voxel coords ----
    k_unq<<<dim3(MAXU / 256), blk, 0, stream>>>(uk, nvox, unq_out);
}
